// TVAdaptor_30829275251372
// MI455X (gfx1250) — compile-verified, matrix operand order checked by reading, not run on hardware
//
#include <hip/hip_runtime.h>

// ---------------------------------------------------------------------------
// TVAdaptor fused style-attention for gfx1250 (MI455X, wave32, WMMA bf16)
//
// B=16, C=256, H=80, W=256 (HW=20480), T=128 (129 tokens, padded to 160)
// Pipeline:
//   k0: one-shot Wq/Wl fp32 -> bf16 conversion (kills per-block cvt chains)
//   k1: per-(b,c) instance-norm stats (mean, rstd)
//   k2: K/V style projections -> bf16; K stored [t][d], V stored transposed
//       [d][t] so all WMMA B-fragments are contiguous 16-byte loads
//   k3: fused per 32-pixel tile (two M=16 halves sharing each B fragment):
//       norm -> q=xn*Wq^T -> logits=q*K^T -> softmax -> out=P*V ->
//       y=out*Wl^T fused with (x+y)*x_mask epilogue straight from accums
// All GEMM stages use v_wmma_f32_16x16x32_bf16.
// ---------------------------------------------------------------------------

typedef __bf16 bf16;
typedef __attribute__((ext_vector_type(16))) __bf16 v16bf;
typedef __attribute__((ext_vector_type(8)))  __bf16 v8bf;
typedef __attribute__((ext_vector_type(8)))  float  v8f;

#define C_DIM  256
#define HW_DIM (80 * 256)
#define W_DIM  256
#define T_DIM  128
#define TP_DIM 160           // padded token count (5 chunks of K=32)
#define M_TILE 32            // pixels per block (2 WMMA M-halves)

// ---- WMMA fragment helpers (ISA 7.12.2 layouts, wave32) -------------------
// 16-bit A matrix 16x32: lanes 0-15 hold M=0..15 / K {k0+0..7, k0+16..23},
//                        lanes 16-31 hold M=0..15 / K {k0+8..15, k0+24..31}.
// All fragment loads are two contiguous 16B chunks (ds/global _b128).

__device__ __forceinline__ v16bf combine8(v8bf lo, v8bf hi) {
    v16bf r;
#pragma unroll
    for (int j = 0; j < 8; ++j) { r[j] = lo[j]; r[8 + j] = hi[j]; }
    return r;
}

__device__ __forceinline__ v16bf load_a_frag(const bf16* lds, int ld, int k0) {
    const int lane = threadIdx.x & 31;
    const int half = lane >> 4;
    const bf16* row = lds + (lane & 15) * ld;
    v8bf lo = *(const v8bf*)(row + k0 + (half ? 8 : 0));
    v8bf hi = *(const v8bf*)(row + k0 + (half ? 24 : 16));
    return combine8(lo, hi);
}

// B matrix 32x16 (K x N), column-major per lane: lane n holds column n0+n;
// source column n is a contiguous row of the row-major bf16 source.
__device__ __forceinline__ v16bf load_b_frag(const bf16* src, int ld,
                                             int n0, int k0) {
    const int lane = threadIdx.x & 31;
    const int half = lane >> 4;
    const bf16* row = src + (size_t)(n0 + (lane & 15)) * ld;
    v8bf lo = *(const v8bf*)(row + k0 + (half ? 8 : 0));
    v8bf hi = *(const v8bf*)(row + k0 + (half ? 24 : 16));
    return combine8(lo, hi);
}

// 32-bit C/D 16x16: VGPR i, lane -> M = i + 8*(lane/16), N = n0 + lane%16.
__device__ __forceinline__ void store_d_bf16(const v8f& d, bf16* lds, int ld, int n0) {
    const int lane = threadIdx.x & 31;
    const int n = n0 + (lane & 15);
    const int half = lane >> 4;
#pragma unroll
    for (int i = 0; i < 8; ++i) lds[(i + 8 * half) * ld + n] = (bf16)d[i];
}

#define WMMA_BF16(a, b, c) \
    __builtin_amdgcn_wmma_f32_16x16x32_bf16(false, (a), false, (b), (short)0, (c), false, false)

// ---------------------------------------------------------------------------
// Kernel 0: one-shot Wq / Wl fp32 -> bf16. grid = (C*C/256), block = 256.
// ---------------------------------------------------------------------------
__global__ __launch_bounds__(256) void tva_cvt_w(const float* __restrict__ Wq,
                                                 const float* __restrict__ Wl,
                                                 bf16* __restrict__ wqbf,
                                                 bf16* __restrict__ wlbf) {
    const int i = blockIdx.x * 256 + threadIdx.x;
    wqbf[i] = (bf16)Wq[i];
    wlbf[i] = (bf16)Wl[i];
}

// ---------------------------------------------------------------------------
// Kernel 1: instance norm stats. grid = (C, B), block = 256 (8 waves).
// ---------------------------------------------------------------------------
__global__ __launch_bounds__(256) void tva_stats(const float* __restrict__ x,
                                                 float* __restrict__ mean,
                                                 float* __restrict__ rstd) {
    const int c = blockIdx.x;
    const int b = blockIdx.y;
    const float* row = x + ((size_t)(b * C_DIM + c)) * HW_DIM;
    float s1 = 0.f, s2 = 0.f;
    for (int i = threadIdx.x; i < HW_DIM; i += 256) {
        float v = row[i];
        s1 += v;
        s2 += v * v;
    }
#pragma unroll
    for (int off = 16; off > 0; off >>= 1) {
        s1 += __shfl_xor(s1, off, 32);
        s2 += __shfl_xor(s2, off, 32);
    }
    __shared__ float r1[8], r2[8];
    const int wave = threadIdx.x >> 5;
    if ((threadIdx.x & 31) == 0) { r1[wave] = s1; r2[wave] = s2; }
    __syncthreads();
    if (threadIdx.x == 0) {
        float a = 0.f, bb = 0.f;
#pragma unroll
        for (int i = 0; i < 8; ++i) { a += r1[i]; bb += r2[i]; }
        const float inv_n = 1.0f / (float)HW_DIM;
        float mu = a * inv_n;
        float var = bb * inv_n - mu * mu;
        mean[b * C_DIM + c] = mu;
        rstd[b * C_DIM + c] = rsqrtf(var + 1e-5f);
    }
}

// ---------------------------------------------------------------------------
// Kernel 2: K/V projections of [time ; sty], padded to TP_DIM rows (zeros).
// K stored [b][t][d] (rows over d); V stored transposed [b][d][t] (rows over
// t) so both serve contiguous WMMA B-fragments.
// grid = (TP_DIM, B), block = 256 (one output channel per thread).
// ---------------------------------------------------------------------------
__global__ __launch_bounds__(256) void tva_kv(const float* __restrict__ sty,
                                              const float* __restrict__ time,
                                              const float* __restrict__ Wk,
                                              const float* __restrict__ Wv,
                                              bf16* __restrict__ kbf,
                                              bf16* __restrict__ vbfT) {
    const int t = blockIdx.x;   // 0..TP_DIM-1 ; real tokens are 0..T_DIM
    const int b = blockIdx.y;
    const int d = threadIdx.x;
    __shared__ float s[C_DIM];

    float sv = 0.f;
    if (t == 0)            sv = time[b * C_DIM + threadIdx.x];
    else if (t <= T_DIM)   sv = sty[((size_t)(b * C_DIM + threadIdx.x)) * T_DIM + (t - 1)];
    s[threadIdx.x] = sv;
    __syncthreads();

    float ka = 0.f, va = 0.f;
    if (t <= T_DIM) {
        const float* wkr = Wk + (size_t)d * C_DIM;
        const float* wvr = Wv + (size_t)d * C_DIM;
#pragma unroll 8
        for (int c = 0; c < C_DIM; ++c) {
            float scv = s[c];
            ka = fmaf(scv, wkr[c], ka);
            va = fmaf(scv, wvr[c], va);
        }
    }
    kbf [((size_t)b * TP_DIM + t) * C_DIM + d] = (bf16)ka;
    vbfT[((size_t)b * C_DIM + d) * TP_DIM + t] = (bf16)va;
}

// ---------------------------------------------------------------------------
// Kernel 3: fused norm + q-proj + attention + V + l-proj + residual.
// grid = (HW/32, B), block = 256 (8 waves); tile = 32 pixels x C=256.
// Each wave computes N=16 output tiles for BOTH M-halves, reusing every
// B fragment for two WMMAs.
// ---------------------------------------------------------------------------
__global__ __launch_bounds__(256) void tva_fused(const float* __restrict__ x,
                                                 const float* __restrict__ x_mask,
                                                 const float* __restrict__ sty_mask,
                                                 const bf16*  __restrict__ wqbf,
                                                 const bf16*  __restrict__ wlbf,
                                                 const float* __restrict__ mean,
                                                 const float* __restrict__ rstd,
                                                 const bf16*  __restrict__ kbf,
                                                 const bf16*  __restrict__ vbfT,
                                                 float* __restrict__ outp) {
    const int tile = blockIdx.x;           // 0..639
    const int b    = blockIdx.y;
    const int p0   = tile * M_TILE;        // 32 consecutive pixels, same h-row
    const int tid  = threadIdx.x;
    const int wave = tid >> 5;
    const int lane = tid & 31;
    const int half = lane >> 4;

    __shared__ bf16  s_xn[M_TILE * C_DIM];   // normalized x tile, [p][c] 16KB
    __shared__ bf16  s_q [M_TILE * C_DIM];   // q tile (1/sqrt(C) folded) 16KB
    __shared__ float s_lg[M_TILE * TP_DIM];  // attention logits          20KB
    __shared__ bf16  s_p [M_TILE * TP_DIM];  // softmax probs             10KB
    __shared__ bf16  s_o [M_TILE * C_DIM];   // attn output tile          16KB

    // -- 1. load + instance-normalize x tile (thread == channel) ------------
    {
        const int c = tid;
        const float* xrow = x + ((size_t)(b * C_DIM + c)) * HW_DIM + p0;
        const float mu = mean[b * C_DIM + c];
        const float rs = rstd[b * C_DIM + c];
#pragma unroll
        for (int p4 = 0; p4 < M_TILE / 4; ++p4) {
            float4 v = ((const float4*)xrow)[p4];
            s_xn[(p4 * 4 + 0) * C_DIM + c] = (bf16)((v.x - mu) * rs);
            s_xn[(p4 * 4 + 1) * C_DIM + c] = (bf16)((v.y - mu) * rs);
            s_xn[(p4 * 4 + 2) * C_DIM + c] = (bf16)((v.z - mu) * rs);
            s_xn[(p4 * 4 + 3) * C_DIM + c] = (bf16)((v.w - mu) * rs);
        }
    }
    __syncthreads();

    // -- 2. q = xn * Wq^T, scaled by 1/sqrt(C) = 1/16 ------------------------
    for (int nt = wave; nt < 16; nt += 8) {
        v8f acc0 = {}, acc1 = {};
#pragma unroll
        for (int kc = 0; kc < 8; ++kc) {
            __builtin_prefetch(wqbf + (size_t)(nt * 16 + (lane & 15)) * C_DIM +
                               ((kc + 1) & 7) * 32, 0, 1);
            v16bf bb = load_b_frag(wqbf, C_DIM, nt * 16, kc * 32);
            v16bf a0 = load_a_frag(s_xn,                 C_DIM, kc * 32);
            v16bf a1 = load_a_frag(s_xn + 16 * C_DIM,    C_DIM, kc * 32);
            acc0 = WMMA_BF16(a0, bb, acc0);
            acc1 = WMMA_BF16(a1, bb, acc1);
        }
#pragma unroll
        for (int i = 0; i < 8; ++i) { acc0[i] *= 0.0625f; acc1[i] *= 0.0625f; }
        store_d_bf16(acc0, s_q,               C_DIM, nt * 16);
        store_d_bf16(acc1, s_q + 16 * C_DIM,  C_DIM, nt * 16);
    }
    __syncthreads();

    // -- 3. logits = q * K^T, with mask / padding ---------------------------
    const bf16* kb = kbf + (size_t)b * TP_DIM * C_DIM;
    for (int nt = wave; nt < TP_DIM / 16; nt += 8) {
        v8f acc0 = {}, acc1 = {};
#pragma unroll
        for (int kc = 0; kc < 8; ++kc) {
            v16bf bb = load_b_frag(kb, C_DIM, nt * 16, kc * 32);
            v16bf a0 = load_a_frag(s_q,              C_DIM, kc * 32);
            v16bf a1 = load_a_frag(s_q + 16 * C_DIM, C_DIM, kc * 32);
            acc0 = WMMA_BF16(a0, bb, acc0);
            acc1 = WMMA_BF16(a1, bb, acc1);
        }
        const int t = nt * 16 + (lane & 15);
        float mval;
        if (t == 0)          mval = 1.f;                                // time token
        else if (t <= T_DIM) mval = sty_mask[(size_t)b * T_DIM + (t - 1)];
        else                 mval = -1.f;                               // padded slot
#pragma unroll
        for (int i = 0; i < 8; ++i) {
            float v0 = acc0[i], v1 = acc1[i];
            if (mval < 0.f)        { v0 = -1e30f;    v1 = -1e30f;    }  // exp == 0
            else if (mval == 0.f)  { v0 = -10000.f;  v1 = -10000.f;  }  // masked
            s_lg[(i + 8 * half) * TP_DIM + t]        = v0;
            s_lg[(i + 8 * half + 16) * TP_DIM + t]   = v1;
        }
    }
    __syncthreads();

    // -- 4. softmax over tokens (4 rows per wave, lane-parallel) ------------
    for (int r = wave * 4; r < wave * 4 + 4; ++r) {
        float vals[TP_DIM / 32];
        float mx = -3.0e38f;
#pragma unroll
        for (int j = 0; j < TP_DIM / 32; ++j) {
            vals[j] = s_lg[r * TP_DIM + lane + j * 32];
            mx = fmaxf(mx, vals[j]);
        }
#pragma unroll
        for (int off = 16; off > 0; off >>= 1) mx = fmaxf(mx, __shfl_xor(mx, off, 32));
        float sum = 0.f;
#pragma unroll
        for (int j = 0; j < TP_DIM / 32; ++j) {
            vals[j] = __expf(vals[j] - mx);
            sum += vals[j];
        }
#pragma unroll
        for (int off = 16; off > 0; off >>= 1) sum += __shfl_xor(sum, off, 32);
        const float inv = 1.0f / sum;
#pragma unroll
        for (int j = 0; j < TP_DIM / 32; ++j)
            s_p[r * TP_DIM + lane + j * 32] = (bf16)(vals[j] * inv);
    }
    __syncthreads();

    // -- 5. out = P * V (V transposed: rows over t are contiguous) ----------
    const bf16* vbT = vbfT + (size_t)b * C_DIM * TP_DIM;
    for (int nt = wave; nt < 16; nt += 8) {
        v8f acc0 = {}, acc1 = {};
#pragma unroll
        for (int kc = 0; kc < TP_DIM / 32; ++kc) {
            v16bf bb = load_b_frag(vbT, TP_DIM, nt * 16, kc * 32);
            v16bf a0 = load_a_frag(s_p,               TP_DIM, kc * 32);
            v16bf a1 = load_a_frag(s_p + 16 * TP_DIM, TP_DIM, kc * 32);
            acc0 = WMMA_BF16(a0, bb, acc0);
            acc1 = WMMA_BF16(a1, bb, acc1);
        }
        store_d_bf16(acc0, s_o,              C_DIM, nt * 16);
        store_d_bf16(acc1, s_o + 16 * C_DIM, C_DIM, nt * 16);
    }
    __syncthreads();

    // -- 6. y = out * Wl^T, fused residual + x_mask epilogue ----------------
    // D-tile rows map to 8 contiguous pixels per lane, so add x / apply mask
    // / store straight from the accumulators (no LDS round-trip).
    const int w0 = p0 % W_DIM;                 // tile never crosses an h-row
    for (int nt = wave; nt < 16; nt += 8) {
        v8f acc0 = {}, acc1 = {};
#pragma unroll
        for (int kc = 0; kc < 8; ++kc) {
            v16bf bb = load_b_frag(wlbf, C_DIM, nt * 16, kc * 32);
            v16bf a0 = load_a_frag(s_o,              C_DIM, kc * 32);
            v16bf a1 = load_a_frag(s_o + 16 * C_DIM, C_DIM, kc * 32);
            acc0 = WMMA_BF16(a0, bb, acc0);
            acc1 = WMMA_BF16(a1, bb, acc1);
        }
        const int cch = nt * 16 + (lane & 15);
#pragma unroll
        for (int mh = 0; mh < 2; ++mh) {
            const v8f& acc = mh ? acc1 : acc0;
            const int prow = 8 * half + 16 * mh;       // pixel base (i = 0..7)
            const size_t xo = ((size_t)(b * C_DIM + cch)) * HW_DIM + p0 + prow;
            const float* xp = x + xo;
            float*       op = outp + xo;
            const float* xm = x_mask + (size_t)b * W_DIM + w0 + prow;
#pragma unroll
            for (int q4 = 0; q4 < 2; ++q4) {
                float4 xv = ((const float4*)xp)[q4];
                float4 mv = ((const float4*)xm)[q4];
                float4 ov;
                ov.x = (xv.x + acc[q4 * 4 + 0]) * mv.x;
                ov.y = (xv.y + acc[q4 * 4 + 1]) * mv.y;
                ov.z = (xv.z + acc[q4 * 4 + 2]) * mv.z;
                ov.w = (xv.w + acc[q4 * 4 + 3]) * mv.w;
                ((float4*)op)[q4] = ov;
            }
        }
    }
}

// ---------------------------------------------------------------------------
extern "C" void kernel_launch(void* const* d_in, const int* in_sizes, int n_in,
                              void* d_out, int out_size, void* d_ws, size_t ws_size,
                              hipStream_t stream) {
    const float* x        = (const float*)d_in[0];
    const float* x_mask   = (const float*)d_in[1];
    const float* sty      = (const float*)d_in[2];
    const float* sty_mask = (const float*)d_in[3];
    const float* time     = (const float*)d_in[4];
    const float* Wq       = (const float*)d_in[5];
    const float* Wk       = (const float*)d_in[6];
    const float* Wv       = (const float*)d_in[7];
    const float* Wl       = (const float*)d_in[8];
    float* outp           = (float*)d_out;

    const int B = 16;
    // workspace layout (all chunks 16B-aligned)
    float* mean = (float*)d_ws;                                  // B*C f32
    float* rstd = mean + (size_t)B * C_DIM;                      // B*C f32
    bf16*  wqbf = (bf16*)(rstd + (size_t)B * C_DIM);             // C*C bf16
    bf16*  wlbf = wqbf + (size_t)C_DIM * C_DIM;                  // C*C bf16
    bf16*  kbf  = wlbf + (size_t)C_DIM * C_DIM;                  // B*Tp*C bf16
    bf16*  vbfT = kbf + (size_t)B * TP_DIM * C_DIM;              // B*C*Tp bf16

    tva_cvt_w<<<dim3(C_DIM * C_DIM / 256), 256, 0, stream>>>(Wq, Wl, wqbf, wlbf);
    tva_stats<<<dim3(C_DIM, B), 256, 0, stream>>>(x, mean, rstd);
    tva_kv<<<dim3(TP_DIM, B), 256, 0, stream>>>(sty, time, Wk, Wv, kbf, vbfT);
    tva_fused<<<dim3(HW_DIM / M_TILE, B), 256, 0, stream>>>(
        x, x_mask, sty_mask, wqbf, wlbf, mean, rstd, kbf, vbfT, outp);
}
